// MLA_68393059222063
// MI455X (gfx1250) — matrix operand order repaired and checked by reading, not compiled, not run on hardware
//
#include <hip/hip_runtime.h>

// ---------------------------------------------------------------------------
// MLA prefill (DeepSeek-style) for gfx1250: bf16 WMMA everywhere, f32 accum.
// GEMM tiles staged by the Tensor Data Mover (tensor_load_to_lds, TENSORcnt),
// double-buffered so DMA overlaps the WMMA chain.
// B=1, S=2048, DIM=2048, H=16, NOPE=128, ROPE=64, VD=128, LORA=512
// ---------------------------------------------------------------------------

#define S_LEN 2048
#define DIMX  2048
#define HEADS 16
#define NOPE  128
#define ROPE  64
#define VD    128
#define LORA  512
#define QKH   (NOPE + ROPE)      // 192
#define KDIM  (LORA + ROPE)      // 576
#define SCALE 0.07216878364870323f  // 192^-0.5

typedef __attribute__((ext_vector_type(16))) __bf16  v16bf;
typedef __attribute__((ext_vector_type(8)))  float   v8f;
typedef __attribute__((ext_vector_type(8)))  unsigned int v8u;
typedef __attribute__((ext_vector_type(4)))  unsigned int tdm_g0_t;
typedef __attribute__((ext_vector_type(8)))  unsigned int tdm_g1_t;
typedef unsigned short ushort_t;

__device__ __forceinline__ unsigned short f2bf(float f) {
    unsigned u = __builtin_bit_cast(unsigned, f);
    return (unsigned short)((u + 0x7fffu + ((u >> 16) & 1u)) >> 16);
}

__device__ __forceinline__ v8f v8f_zero() {
    v8f x;
#pragma unroll
    for (int i = 0; i < 8; ++i) x[i] = 0.0f;
    return x;
}

// A-operand fragment (16x32 bf16). p points at the start of the lane's row
// (already offset by this k-step). Layout: lanes<16 hold K {0..15},
// lanes>=16 hold K {16..31}, packed 2 per VGPR.
__device__ __forceinline__ v16bf frag_a(const ushort_t* p, int lane) {
    int kb = (lane & 16) ? 16 : 0;
    v8u u;
#pragma unroll
    for (int j = 0; j < 8; ++j) {
        int k = kb + 2 * j;
        u[j] = *(const unsigned int*)(p + k);
    }
    return __builtin_bit_cast(v16bf, u);
}

// B-operand fragment (32x16 bf16). p points at the lane's column base with
// the K half-offset (0 for lanes<16, 16 for lanes>=16) already applied; the
// 8 dwords are then contiguous K pairs.
__device__ __forceinline__ v16bf frag_b(const ushort_t* p) {
    v8u u;
#pragma unroll
    for (int j = 0; j < 8; ++j) u[j] = *(const unsigned int*)(p + 2 * j);
    return __builtin_bit_cast(v16bf, u);
}

#define WMMA_BF16(a, b, c)                                                     \
    __builtin_amdgcn_wmma_f32_16x16x32_bf16(false, (a), false, (b), (short)0,  \
                                            (c), false, false)

// ---------------------------------------------------------------------------
// Tensor Data Mover: DMA a rows x cols (bf16) tile, row stride = stride elems,
// from global into LDS (packed row-major, row pitch = cols). Issued by one
// wave; completion tracked with TENSORcnt.
// D# layout per CDNA5 ISA 8.3/8.4 (group0: count/lds/global/type,
// group1: data_size + tensor/tile dims + dim0 stride).
// ---------------------------------------------------------------------------
__device__ __forceinline__ void tdm_load_tile(const ushort_t* gsrc,
                                              unsigned lds_off, int cols,
                                              int rows, long stride) {
    unsigned long long ga = (unsigned long long)(size_t)gsrc;
    tdm_g0_t g0;
    g0[0] = 1u;                                          // count=1, user mode
    g0[1] = lds_off;                                     // LDS byte address
    g0[2] = (unsigned)ga;                                // global addr [31:0]
    g0[3] = (unsigned)((ga >> 32) & 0x01FFFFFFu) | 0x80000000u; // [56:32]|type=2

    unsigned tdim0 = (unsigned)stride;   // tensor row length (elems)
    unsigned tdim1 = (unsigned)rows;     // rows available
    unsigned sl = (unsigned)stride;
    unsigned sh = (unsigned)(((unsigned long long)stride >> 32) & 0xFFFFu);
    tdm_g1_t g1;
    g1[0] = 0x00010000u;                                  // data_size=2B
    g1[1] = (tdim0 & 0xFFFFu) << 16;                      // tensor_dim0 lo
    g1[2] = ((tdim0 >> 16) & 0xFFFFu) | (tdim1 << 16);    // dim0 hi | dim1 lo
    g1[3] = ((tdim1 >> 16) & 0xFFFFu) |
            (((unsigned)cols & 0xFFFFu) << 16);           // dim1 hi | tile_dim0
    g1[4] = (unsigned)rows & 0xFFFFu;                     // tile_dim1 (dim2=0)
    g1[5] = sl;                                           // dim0_stride [31:0]
    g1[6] = sh;                                           // dim0_stride [47:32]
    g1[7] = 0u;                                           // dim1_stride unused
    asm volatile("tensor_load_to_lds %0, %1" ::"s"(g0), "s"(g1) : "memory");
}

// ---------------------------------------------------------------------------
// Elementwise fp32 -> bf16
// ---------------------------------------------------------------------------
__global__ void cvt_bf16(const float* __restrict__ src, ushort_t* __restrict__ dst,
                         long n) {
    long i = (long)blockIdx.x * 256 + threadIdx.x;
    if (i < n) dst[i] = f2bf(src[i]);
}

// Split wkv_b [H][256][512] into w_ukT [H][512c][128d] and w_uv [H][128d][512c]
__global__ void prep_wuk_wuv(const float* __restrict__ wkvb,
                             ushort_t* __restrict__ wukT,
                             ushort_t* __restrict__ wuv) {
    long i = (long)blockIdx.x * 256 + threadIdx.x;
    if (i >= (long)HEADS * 256 * LORA) return;
    int c = (int)(i % LORA);
    long t = i / LORA;
    int dv = (int)(t % 256);
    int h  = (int)(t / 256);
    unsigned short v = f2bf(wkvb[i]);
    if (dv < NOPE)
        wukT[((long)h * LORA + c) * NOPE + dv] = v;
    else
        wuv[((long)h * VD + (dv - NOPE)) * LORA + c] = v;
}

// ---------------------------------------------------------------------------
// Generic C[M,N] = A[M,K] * W[N,K]^T  (bf16 in, f32 accumulate)
// tile: 128(M) x 64(N), 256 threads = 8 waves, each wave owns a 16-row strip.
// A/W tiles staged by the TDM, double-buffered: DMA of tile i+1 overlaps the
// WMMA chain of tile i; sync via s_wait_tensorcnt + workgroup barrier.
// ---------------------------------------------------------------------------
template <bool BF16OUT>
__global__ __launch_bounds__(256) void gemm_xwT(const ushort_t* __restrict__ A,
                                                const ushort_t* __restrict__ W,
                                                void* __restrict__ Cv, int M,
                                                int N, int K, long sA, long sW,
                                                long sC, int ldc) {
    __shared__ __align__(16) ushort_t lA[2][128 * 32];
    __shared__ __align__(16) ushort_t lB[2][64 * 32];

    int b  = blockIdx.z;
    const ushort_t* Ab = A + (long)b * sA;
    const ushort_t* Wb = W + (long)b * sW;
    int m0 = blockIdx.y * 128, n0 = blockIdx.x * 64;
    int tid = threadIdx.x, lane = tid & 31, wave = tid >> 5;

    v8f acc[4];
#pragma unroll
    for (int nt = 0; nt < 4; ++nt) acc[nt] = v8f_zero();

    int nk = K >> 5;  // K / 32
    if (tid < 32) {   // wave 0 drives the DMA engine
        tdm_load_tile(Ab + (long)m0 * K, (unsigned)(size_t)&lA[0][0], 32, 128, K);
        tdm_load_tile(Wb + (long)n0 * K, (unsigned)(size_t)&lB[0][0], 32, 64, K);
    }

    for (int i = 0; i < nk; ++i) {
        int kk = i * 32;
        if (tid < 32) {
            if (i + 1 < nk) {
                tdm_load_tile(Ab + (long)m0 * K + kk + 32,
                              (unsigned)(size_t)&lA[(i + 1) & 1][0], 32, 128, K);
                tdm_load_tile(Wb + (long)n0 * K + kk + 32,
                              (unsigned)(size_t)&lB[(i + 1) & 1][0], 32, 64, K);
                __builtin_amdgcn_s_wait_tensorcnt(2);  // current pair landed
            } else {
                __builtin_amdgcn_s_wait_tensorcnt(0);
            }
        }
        __syncthreads();

        const ushort_t* la = lA[i & 1];
        const ushort_t* lb = lB[i & 1];
        v16bf a = frag_a(la + (wave * 16 + (lane & 15)) * 32, lane);
        int kb = (lane & 16) ? 16 : 0;
#pragma unroll
        for (int nt = 0; nt < 4; ++nt) {
            v16bf bb = frag_b(lb + (nt * 16 + (lane & 15)) * 32 + kb);
            acc[nt] = WMMA_BF16(a, bb, acc[nt]);
        }
        __syncthreads();
    }

    int rbase = m0 + wave * 16 + ((lane & 16) ? 8 : 0);
#pragma unroll
    for (int nt = 0; nt < 4; ++nt) {
        int col = n0 + nt * 16 + (lane & 15);
#pragma unroll
        for (int r = 0; r < 8; ++r) {
            long off = (long)b * sC + (long)(rbase + r) * ldc + col;
            if (BF16OUT)
                ((ushort_t*)Cv)[off] = f2bf(acc[nt][r]);
            else
                ((float*)Cv)[off] = acc[nt][r];
        }
    }
}

// ---------------------------------------------------------------------------
// RoPE on q_pe; split q into per-head bf16 buffers qn [H][S][128], qpe [H][S][64]
// ---------------------------------------------------------------------------
__global__ void prep_q(const float* __restrict__ q, const float* __restrict__ cosT,
                       const float* __restrict__ sinT, ushort_t* __restrict__ qn,
                       ushort_t* __restrict__ qpe) {
    long t = (long)blockIdx.x * 256 + threadIdx.x;  // [0, S*H)
    int s = (int)(t >> 4);
    int h = (int)(t & 15);
    const float* qr = q + (long)s * (HEADS * QKH) + h * QKH;
    ushort_t* qno = qn + ((long)h * S_LEN + s) * NOPE;
#pragma unroll 4
    for (int d = 0; d < NOPE; ++d) qno[d] = f2bf(qr[d]);
    ushort_t* qp = qpe + ((long)h * S_LEN + s) * ROPE;
#pragma unroll 4
    for (int i = 0; i < ROPE / 2; ++i) {
        float x0 = qr[NOPE + 2 * i], x1 = qr[NOPE + 2 * i + 1];
        float c = cosT[s * 32 + i], sn = sinT[s * 32 + i];
        qp[2 * i]     = f2bf(x0 * c - x1 * sn);
        qp[2 * i + 1] = f2bf(x0 * sn + x1 * c);
    }
}

// ---------------------------------------------------------------------------
// RMSNorm kv_c (+store row-major and transposed), RoPE k_pe.  1 block per row.
// ---------------------------------------------------------------------------
__global__ __launch_bounds__(128) void prep_kv(
    const float* __restrict__ kv, const float* __restrict__ normw,
    const float* __restrict__ cosT, const float* __restrict__ sinT,
    ushort_t* __restrict__ kvc, ushort_t* __restrict__ kvcT,
    ushort_t* __restrict__ kpe) {
    int s = blockIdx.x, tid = threadIdx.x;
    __shared__ float red[128];
    const float* row = kv + (long)s * KDIM;
    float ss = 0.f;
    for (int c = tid; c < LORA; c += 128) { float v = row[c]; ss += v * v; }
    red[tid] = ss;
    __syncthreads();
    for (int st = 64; st > 0; st >>= 1) {
        if (tid < st) red[tid] += red[tid + st];
        __syncthreads();
    }
    float inv = rsqrtf(red[0] / (float)LORA + 1e-6f);
    for (int c = tid; c < LORA; c += 128) {
        unsigned short b = f2bf(row[c] * inv * normw[c]);
        kvc[(long)s * LORA + c]    = b;
        kvcT[(long)c * S_LEN + s]  = b;
    }
    if (tid < 32) {
        int i = tid;
        float x0 = row[LORA + 2 * i], x1 = row[LORA + 2 * i + 1];
        float c = cosT[s * 32 + i], sn = sinT[s * 32 + i];
        kpe[(long)s * ROPE + 2 * i]     = f2bf(x0 * c - x1 * sn);
        kpe[(long)s * ROPE + 2 * i + 1] = f2bf(x0 * sn + x1 * c);
    }
}

// ---------------------------------------------------------------------------
// Flash attention in latent space. One wave per (head, 16-query tile).
// Q tile (16 x 576 = 512 latent + 64 rope) in LDS; keys processed 32 at a time:
// 2 score fragments (18 WMMAs each K-chain) -> online softmax -> P through LDS
// -> 32 WMMAs of P(16x32) x V(32x16) accumulating O (16x512) in 256 VGPRs.
// ---------------------------------------------------------------------------
__global__ __launch_bounds__(32) void mla_attn(
    const ushort_t* __restrict__ qc, const ushort_t* __restrict__ qpe,
    const ushort_t* __restrict__ kvc, const ushort_t* __restrict__ kvcT,
    const ushort_t* __restrict__ kpe, ushort_t* __restrict__ o_out) {
    __shared__ __align__(16) ushort_t qs[16 * KDIM];  // 18 KB
    __shared__ __align__(8)  ushort_t ps[16 * 32];    // 1 KB

    int h  = blockIdx.y;
    int qb = blockIdx.x * 16;
    int lane = threadIdx.x;
    int ln = lane & 15;
    int hi = (lane & 16) ? 1 : 0;

    // stage Q tile: cols 0..511 = q_c, cols 512..575 = q_pe
    for (int i = lane; i < 16 * KDIM / 8; i += 32) {
        int idx = i * 8;
        int r = idx / KDIM, c = idx % KDIM;
        uint4 v;
        if (c < LORA)
            v = *(const uint4*)&qc[((long)h * S_LEN + qb + r) * LORA + c];
        else
            v = *(const uint4*)&qpe[((long)h * S_LEN + qb + r) * ROPE + (c - LORA)];
        *(uint4*)&qs[idx] = v;
    }

    float m_i[8], l_i[8];
#pragma unroll
    for (int r = 0; r < 8; ++r) { m_i[r] = -1e30f; l_i[r] = 0.f; }
    v8f acc[32];
#pragma unroll
    for (int t = 0; t < 32; ++t) acc[t] = v8f_zero();

    for (int tb = 0; tb < qb + 16; tb += 32) {
        // ---- scores: 16 queries x 32 keys over K = 512 + 64 ----
        v8f s0 = v8f_zero(), s1 = v8f_zero();
        int kbB = (lane & 16) ? 16 : 0;
#pragma unroll
        for (int kk = 0; kk < 18; ++kk) {
            const ushort_t *b0p, *b1p;
            if (kk < 16) {
                b0p = kvc + ((long)(tb + ln)) * LORA + kk * 32 + kbB;
                b1p = kvc + ((long)(tb + 16 + ln)) * LORA + kk * 32 + kbB;
            } else {
                b0p = kpe + ((long)(tb + ln)) * ROPE + (kk - 16) * 32 + kbB;
                b1p = kpe + ((long)(tb + 16 + ln)) * ROPE + (kk - 16) * 32 + kbB;
            }
            v16bf a = frag_a(qs + ln * KDIM + kk * 32, lane);
            s0 = WMMA_BF16(a, frag_b(b0p), s0);
            s1 = WMMA_BF16(a, frag_b(b1p), s1);
        }

        // ---- online softmax (rows live on 16-lane halves) ----
#pragma unroll
        for (int r = 0; r < 8; ++r) {
            int grow = qb + r + hi * 8;
            float v0 = s0[r] * SCALE + (((tb + ln) <= grow) ? 0.f : -1e30f);
            float v1 = s1[r] * SCALE + (((tb + 16 + ln) <= grow) ? 0.f : -1e30f);
            float mx = fmaxf(v0, v1);
            mx = fmaxf(mx, __shfl_xor(mx, 1, 32));
            mx = fmaxf(mx, __shfl_xor(mx, 2, 32));
            mx = fmaxf(mx, __shfl_xor(mx, 4, 32));
            mx = fmaxf(mx, __shfl_xor(mx, 8, 32));
            float mn = fmaxf(m_i[r], mx);
            float p0 = __expf(v0 - mn), p1 = __expf(v1 - mn);
            float rs = __expf(m_i[r] - mn);
            float sum = p0 + p1;
            sum += __shfl_xor(sum, 1, 32);
            sum += __shfl_xor(sum, 2, 32);
            sum += __shfl_xor(sum, 4, 32);
            sum += __shfl_xor(sum, 8, 32);
            l_i[r] = l_i[r] * rs + sum;
            m_i[r] = mn;
            ps[(r + hi * 8) * 32 + ln]      = f2bf(p0);
            ps[(r + hi * 8) * 32 + ln + 16] = f2bf(p1);
#pragma unroll
            for (int t = 0; t < 32; ++t) acc[t][r] *= rs;
        }

        // ---- O += P(16x32) * V(32x16 per c-tile), V from kv_c^T ----
        v16bf ap = frag_a(ps + ln * 32, lane);
#pragma unroll
        for (int ct = 0; ct < 32; ++ct) {
            const ushort_t* bp =
                kvcT + ((long)(ct * 16 + ln)) * S_LEN + tb + kbB;
            acc[ct] = WMMA_BF16(ap, frag_b(bp), acc[ct]);
        }
    }

    // ---- normalize and write o [H][S][512] bf16 ----
#pragma unroll
    for (int r = 0; r < 8; ++r) {
        float inv = 1.0f / l_i[r];
        int grow = qb + r + hi * 8;
#pragma unroll
        for (int ct = 0; ct < 32; ++ct)
            o_out[((long)h * S_LEN + grow) * LORA + ct * 16 + ln] =
                f2bf(acc[ct][r] * inv);
    }
}

// ---------------------------------------------------------------------------
// Host-side orchestration
// ---------------------------------------------------------------------------
extern "C" void kernel_launch(void* const* d_in, const int* in_sizes, int n_in,
                              void* d_out, int out_size, void* d_ws,
                              size_t ws_size, hipStream_t stream) {
    const float* x     = (const float*)d_in[0];
    const float* cosT  = (const float*)d_in[1];
    const float* sinT  = (const float*)d_in[2];
    const float* wq    = (const float*)d_in[3];
    const float* wkv_a = (const float*)d_in[4];
    const float* kvnw  = (const float*)d_in[5];
    const float* wkv_b = (const float*)d_in[6];
    const float* wo    = (const float*)d_in[7];
    (void)in_sizes; (void)n_in; (void)out_size; (void)ws_size;

    char* ws = (char*)d_ws;
    size_t off = 0;
    auto alloc = [&](size_t bytes) -> char* {
        char* p = ws + off;
        off = (off + bytes + 255) & ~(size_t)255;
        return p;
    };

    ushort_t* x_bf    = (ushort_t*)alloc((size_t)S_LEN * DIMX * 2);
    ushort_t* wq_bf   = (ushort_t*)alloc((size_t)HEADS * QKH * DIMX * 2);
    ushort_t* wkva_bf = (ushort_t*)alloc((size_t)KDIM * DIMX * 2);
    ushort_t* wo_bf   = (ushort_t*)alloc((size_t)DIMX * HEADS * VD * 2);
    ushort_t* wukT_bf = (ushort_t*)alloc((size_t)HEADS * LORA * NOPE * 2);
    ushort_t* wuv_bf  = (ushort_t*)alloc((size_t)HEADS * VD * LORA * 2);
    float*    q_f32   = (float*)alloc((size_t)S_LEN * HEADS * QKH * 4);
    float*    kv_f32  = (float*)alloc((size_t)S_LEN * KDIM * 4);
    ushort_t* qn_bf   = (ushort_t*)alloc((size_t)HEADS * S_LEN * NOPE * 2);
    ushort_t* qpe_bf  = (ushort_t*)alloc((size_t)HEADS * S_LEN * ROPE * 2);
    ushort_t* qc_bf   = (ushort_t*)alloc((size_t)HEADS * S_LEN * LORA * 2);
    ushort_t* kvc_bf  = (ushort_t*)alloc((size_t)S_LEN * LORA * 2);
    ushort_t* kvcT_bf = (ushort_t*)alloc((size_t)LORA * S_LEN * 2);
    ushort_t* kpe_bf  = (ushort_t*)alloc((size_t)S_LEN * ROPE * 2);
    ushort_t* o_bf    = (ushort_t*)alloc((size_t)HEADS * S_LEN * LORA * 2);
    ushort_t* oprj_bf = (ushort_t*)alloc((size_t)S_LEN * HEADS * VD * 2);

    auto cvt = [&](const float* s, ushort_t* d, long n) {
        cvt_bf16<<<dim3((unsigned)((n + 255) / 256)), dim3(256), 0, stream>>>(s, d, n);
    };
    cvt(x, x_bf, (long)S_LEN * DIMX);
    cvt(wq, wq_bf, (long)HEADS * QKH * DIMX);
    cvt(wkv_a, wkva_bf, (long)KDIM * DIMX);
    cvt(wo, wo_bf, (long)DIMX * HEADS * VD);
    prep_wuk_wuv<<<dim3((HEADS * 256 * LORA + 255) / 256), dim3(256), 0, stream>>>(
        wkv_b, wukT_bf, wuv_bf);

    // q = x @ wq^T  -> fp32 [S, 3072]
    gemm_xwT<false><<<dim3(HEADS * QKH / 64, S_LEN / 128, 1), dim3(256), 0, stream>>>(
        x_bf, wq_bf, q_f32, S_LEN, HEADS * QKH, DIMX, 0, 0, 0, HEADS * QKH);
    // kv = x @ wkv_a^T -> fp32 [S, 576]
    gemm_xwT<false><<<dim3(KDIM / 64, S_LEN / 128, 1), dim3(256), 0, stream>>>(
        x_bf, wkva_bf, kv_f32, S_LEN, KDIM, DIMX, 0, 0, 0, KDIM);

    prep_q<<<dim3(S_LEN * HEADS / 256), dim3(256), 0, stream>>>(q_f32, cosT, sinT,
                                                                qn_bf, qpe_bf);
    prep_kv<<<dim3(S_LEN), dim3(128), 0, stream>>>(kv_f32, kvnw, cosT, sinT,
                                                   kvc_bf, kvcT_bf, kpe_bf);

    // q_c[h] = q_nope[h] @ w_uk[h]  (A:[S,128], W:[512,128]) -> bf16 [H][S][512]
    gemm_xwT<true><<<dim3(LORA / 64, S_LEN / 128, HEADS), dim3(256), 0, stream>>>(
        qn_bf, wukT_bf, qc_bf, S_LEN, LORA, NOPE, (long)S_LEN * NOPE,
        (long)LORA * NOPE, (long)S_LEN * LORA, LORA);

    // attention
    mla_attn<<<dim3(S_LEN / 16, HEADS), dim3(32), 0, stream>>>(
        qc_bf, qpe_bf, kvc_bf, kvcT_bf, kpe_bf, o_bf);

    // o_proj[:, h*128:(h+1)*128] = o[h] @ w_uv[h]^T  -> bf16 [S, 2048]
    gemm_xwT<true><<<dim3(VD / 64, S_LEN / 128, HEADS), dim3(256), 0, stream>>>(
        o_bf, wuv_bf, oprj_bf, S_LEN, VD, LORA, (long)S_LEN * LORA,
        (long)VD * LORA, (long)VD, HEADS * VD);

    // out = o_proj @ wo^T -> fp32 [S, 2048]
    gemm_xwT<false><<<dim3(DIMX / 64, S_LEN / 128, 1), dim3(256), 0, stream>>>(
        oprj_bf, wo_bf, (float*)d_out, S_LEN, DIMX, HEADS * VD, 0, 0, 0, DIMX);
}
